// VAE_smearing_34505767256328
// MI455X (gfx1250) — matrix operand order repaired and checked by reading, not compiled, not run on hardware
//
#include <hip/hip_runtime.h>
#include <hip/hip_bf16.h>
#include <math.h>

#define N_TYPES 5
#define N_ETYPES 25
#define NN 4096
#define EE 8192
#define HH 128
#define EMBD 64
#define EDIM 4
#define TDIM 3
#define ZDIM 3

typedef __attribute__((ext_vector_type(16))) __bf16 v16bf;
typedef __attribute__((ext_vector_type(8)))  float  v8f;

union BFrag { v16bf v; uint4 q[2]; unsigned short u[16]; };

__device__ __forceinline__ unsigned short f2bf(float f) {
  unsigned int u = __float_as_uint(f);
  u += 0x7FFFu + ((u >> 16) & 1u);            // round to nearest even
  return (unsigned short)(u >> 16);
}
__device__ __forceinline__ float bf2f(unsigned short u) {
  return __uint_as_float(((unsigned int)u) << 16);
}
__device__ __forceinline__ float lrelu(float x, float s) { return x > 0.f ? x : s * x; }

// ---------------------------------------------------------------------------
// Batched bf16 WMMA GEMM.  A: bf16 activations (row-major, ld = K).
// WT: bf16 weights TRANSPOSED (t, o, k) so each B fragment is 2 x b128 loads.
// Each wave computes a 16x64 output strip (4 accumulators, A frag reused 4x).
// epi: 0 = identity, 1 = leaky(0.01), 2 = gated residual g + g*sigmoid(acc).
// Writes f32 C (optional) and bf16 mirror Cbf (optional).
// ---------------------------------------------------------------------------
template<int K, int O>
__global__ __launch_bounds__(256)
void gemm_wmma(const unsigned short* __restrict__ A, const unsigned short* __restrict__ WT,
               const float* __restrict__ bias, const float* __restrict__ gateSrc,
               float* __restrict__ C, unsigned short* __restrict__ Cbf,
               int ldC, int colOff, long aBS, long cBS, int amapDiv, int epi)
{
  constexpr int CT = 4;                       // 16-col tiles per wave
  constexpr int colGroups = O / (16 * CT);    // 2 (O=128) or 1 (O=64)
  constexpr int nTiles = (NN / 16) * colGroups;
  const int lane = threadIdx.x & 31;
  const int wave = threadIdx.x >> 5;
  const int tile = blockIdx.x * 8 + wave;
  if (tile >= nTiles) return;                 // wave-uniform
  const int rowTile = tile / colGroups;
  const int colBase = (tile % colGroups) * (16 * CT);
  const int t  = blockIdx.y;
  const int aT = t / amapDiv;                 // amapDiv=5 implements h[ST]
  const int r  = lane & 15;
  const int kh = lane >> 4;

  const unsigned short* Arow = A + (long)aT * aBS + (long)(rowTile * 16 + r) * K;
  const unsigned short* Wt   = WT + (long)t * O * K;

  const v8f vzero = {0.f, 0.f, 0.f, 0.f, 0.f, 0.f, 0.f, 0.f};
  v8f acc[CT];
#pragma unroll
  for (int c = 0; c < CT; ++c) acc[c] = vzero;

#pragma unroll
  for (int k0 = 0; k0 < K; k0 += 32) {
    BFrag a;
    // A 16x32 bf16 fragment: lane holds K = kh*8+{0..7} and 16+kh*8+{0..7}
    a.q[0] = *(const uint4*)(Arow + k0 + kh * 8);
    a.q[1] = *(const uint4*)(Arow + k0 + 16 + kh * 8);
#pragma unroll
    for (int c = 0; c < CT; ++c) {
      // B 32x16 bf16 fragment: lane holds K = kh*8+{0..7} and 16+kh*8+{0..7} of column col
      const unsigned short* wrow =
          Wt + (long)(colBase + c * 16 + (lane & 15)) * K + k0 + kh * 8;
      BFrag b;
      b.q[0] = *(const uint4*)(wrow);
      b.q[1] = *(const uint4*)(wrow + 16);
      acc[c] = __builtin_amdgcn_wmma_f32_16x16x32_bf16(false, a.v, false, b.v,
                                                       (short)0, acc[c], false, false);
    }
  }

#pragma unroll
  for (int c = 0; c < CT; ++c) {
    const int col = colBase + c * 16 + (lane & 15);
    const float bv = bias ? bias[t * O + col] : 0.f;
#pragma unroll
    for (int v = 0; v < 8; ++v) {
      const int row = rowTile * 16 + v + (kh ? 8 : 0);
      const long cidx = (long)t * cBS + (long)row * ldC + colOff + col;
      float x = acc[c][v] + bv;
      float outv;
      if (epi == 1)      outv = lrelu(x, 0.01f);
      else if (epi == 2) { float g = gateSrc[cidx]; outv = g + g * (1.f / (1.f + __expf(-x))); }
      else               outv = x;
      if (C)   C[cidx]   = outv;
      if (Cbf) Cbf[cidx] = f2bf(outv);
    }
  }
}

// ---------------------------------------------------------------------------
// Weight convert+transpose: WT[t][o*K+k] = bf16(W[t][k*O+o])
__global__ void wconv_kernel(const float* __restrict__ W, unsigned short* __restrict__ WT,
                             int K, int O, long n) {
  long i = (long)blockIdx.x * blockDim.x + threadIdx.x;
  if (i >= n) return;
  long ko = i % (long)(K * O); int t = i / (K * O);
  int o = ko / K, k = ko % K;
  WT[(long)t * K * O + (long)o * K + k] = f2bf(W[(long)t * K * O + (long)k * O + o]);
}

__global__ void fill_kernel(float* p, float v, long n) {
  long i = (long)blockIdx.x * blockDim.x + threadIdx.x;
  if (i < n) p[i] = v;
}

// embed (K=3): writes bf16 activations directly (only the self-GEMM consumes it)
__global__ void emb_kernel(const float* __restrict__ x, const float* __restrict__ Wp,
                           const float* __restrict__ bp, unsigned short* __restrict__ h,
                           int inDim) {
  long i = (long)blockIdx.x * blockDim.x + threadIdx.x;
  if (i >= (long)N_TYPES * NN * EMBD) return;
  int o = i % EMBD; long tn = i / EMBD; int n = tn % NN; int t = tn / NN;
  const float* xr = x + ((long)t * NN + n) * inDim;
  const float* Wt = Wp + (long)t * inDim * EMBD;
  float s = bp[t * EMBD + o];
  for (int k = 0; k < inDim; ++k) s += xr[k] * Wt[k * EMBD + o];
  h[((long)t * NN + n) * EMBD + o] = f2bf(s);
}

// ws = W*att_src, wd = W*att_dst (128 each), we = W_edge*att_edge (4)
__global__ void attvec_kernel(const float* __restrict__ W, const float* __restrict__ attS,
                              const float* __restrict__ attD, const float* __restrict__ We,
                              const float* __restrict__ attE,
                              float* ws, float* wd, float* we) {
  int et = blockIdx.x; int k = threadIdx.x;   // 128 threads
  const float* Wt = W + (long)et * HH * HH;
  float ss = 0.f, sd = 0.f;
  for (int o = 0; o < HH; ++o) {
    float w = Wt[k * HH + o];
    ss += w * attS[et * HH + o];
    sd += w * attD[et * HH + o];
  }
  ws[et * HH + k] = ss; wd[et * HH + k] = sd;
  if (k < EDIM) {
    const float* Wet = We + (long)et * EDIM * HH;
    float se = 0.f;
    for (int o = 0; o < HH; ++o) se += Wet[k * HH + o] * attE[et * HH + o];
    we[et * EDIM + k] = se;
  }
}

__global__ void node_att_kernel(const float* __restrict__ h, const float* __restrict__ ws,
                                const float* __restrict__ wd, float* a_src, float* a_dst) {
  long i = (long)blockIdx.x * blockDim.x + threadIdx.x;
  if (i >= (long)N_ETYPES * NN) return;
  int n = i % NN; int et = i / NN;
  int st = et / N_TYPES, dt = et % N_TYPES;
  const float* hs = h + ((long)st * NN + n) * HH;
  const float* hd = h + ((long)dt * NN + n) * HH;
  float ss = 0.f, sd = 0.f;
  for (int k = 0; k < HH; ++k) { ss += hs[k] * ws[et * HH + k]; sd += hd[k] * wd[et * HH + k]; }
  a_src[i] = ss; a_dst[i] = sd;
}

__device__ __forceinline__ void atomicMaxF(float* addr, float v) {
  if (v >= 0.f) atomicMax((int*)addr, __float_as_int(v));
  else          atomicMin((unsigned int*)addr, (unsigned int)__float_as_int(v));
}

__global__ void edge_alpha_kernel(const int* __restrict__ ei, const float* __restrict__ ea,
                                  const float* __restrict__ we, const float* __restrict__ a_src,
                                  const float* __restrict__ a_dst, float* alpha, float* m) {
  long i = (long)blockIdx.x * blockDim.x + threadIdx.x;
  if (i >= (long)N_ETYPES * EE) return;
  int e = i % EE; int et = i / EE;
  int src = ei[(long)et * 2 * EE + e];
  int dst = ei[(long)et * 2 * EE + EE + e];
  float a = a_src[(long)et * NN + src] + a_dst[(long)et * NN + dst];
  const float* ear = ea + ((long)et * EE + e) * EDIM;
#pragma unroll
  for (int f = 0; f < EDIM; ++f) a += ear[f] * we[et * EDIM + f];
  a = lrelu(a, 0.2f);
  alpha[i] = a;
  atomicMaxF(&m[(long)et * NN + dst], a);
}

__global__ void edge_exp_kernel(const int* __restrict__ ei, const float* __restrict__ alpha,
                                const float* __restrict__ m, float* ex, float* den) {
  long i = (long)blockIdx.x * blockDim.x + threadIdx.x;
  if (i >= (long)N_ETYPES * EE) return;
  int e = i % EE; int et = i / EE;
  int dst = ei[(long)et * 2 * EE + EE + e];
  float v = __expf(alpha[i] - m[(long)et * NN + dst]);
  ex[i] = v;
  atomicAdd(&den[(long)et * NN + dst], v);
}

// 32 threads per edge, 4 channels each: msg[et,dst,:] += bf2f(phbf[et,src,:])*w
__global__ void edge_msg_kernel(const int* __restrict__ ei, const float* __restrict__ ex,
                                const float* __restrict__ den,
                                const unsigned short* __restrict__ phbf, float* msg) {
  long i = (long)blockIdx.x * blockDim.x + threadIdx.x;
  if (i >= (long)N_ETYPES * EE * 32) return;
  int ch = (int)(i & 31); long eg = i >> 5;
  int e = eg % EE; int et = eg / EE;
  int src = ei[(long)et * 2 * EE + e];
  int dst = ei[(long)et * 2 * EE + EE + e];
  float w = ex[eg] / den[(long)et * NN + dst];
  const uint2 pq = *(const uint2*)(phbf + (((long)et * NN + src) * HH) + ch * 4);
  float* md = msg + (((long)et * NN + dst) * HH) + ch * 4;
  atomicAdd(md + 0, __uint_as_float((pq.x & 0xFFFFu) << 16) * w);
  atomicAdd(md + 1, __uint_as_float(pq.x & 0xFFFF0000u) * w);
  atomicAdd(md + 2, __uint_as_float((pq.y & 0xFFFFu) << 16) * w);
  atomicAdd(md + 3, __uint_as_float(pq.y & 0xFFFF0000u) * w);
}

// comm[d,n,c] = leaky(max_s(msg[s*5+d,n,c] + bias)) -> bf16 (only ccomp consumes)
__global__ void comm_kernel(const float* __restrict__ msg, const float* __restrict__ gbias,
                            unsigned short* __restrict__ commbf) {
  long i = (long)blockIdx.x * blockDim.x + threadIdx.x;
  if (i >= (long)N_TYPES * NN * HH) return;
  int c = i % HH; long dn = i / HH; int n = dn % NN; int d = dn / NN;
  float mx = -3.4e38f;
#pragma unroll
  for (int s = 0; s < N_TYPES; ++s) {
    int et = s * N_TYPES + d;
    float v = msg[(((long)et * NN) + n) * HH + c] + gbias[et * HH + c];
    mx = fmaxf(mx, v);
  }
  commbf[(((long)d * NN) + n) * HH + c] = f2bf(lrelu(mx, 0.01f));
}

__global__ void final_kernel(const float* __restrict__ h, const float* __restrict__ W,
                             const float* __restrict__ b, float* out, int O) {
  long i = (long)blockIdx.x * blockDim.x + threadIdx.x;
  if (i >= (long)N_TYPES * NN * O) return;
  int o = i % O; long tn = i / O; int n = tn % NN; int t = tn / NN;
  const float* hr = h + ((long)t * NN + n) * HH;
  const float* Wt = W + (long)t * HH * O;
  float s = b[t * O + o];
  for (int k = 0; k < HH; ++k) s += hr[k] * Wt[k * O + o];
  out[i] = s;
}

__global__ void reparam_kernel(const float* __restrict__ lat, const float* __restrict__ eps,
                               float* z, float* muO, float* lvO) {
  long i = (long)blockIdx.x * blockDim.x + threadIdx.x;
  if (i >= (long)N_TYPES * NN * ZDIM) return;
  int zi = i % ZDIM; long tn = i / ZDIM;
  float mu = lat[tn * 2 * ZDIM + 2 * zi];
  float lv = lat[tn * 2 * ZDIM + 2 * zi + 1];
  z[i] = mu + __expf(0.5f * lv) * eps[i];
  muO[i] = mu; lvO[i] = lv;
}

// ---------------------------------------------------------------------------
// Param leaf offsets (jax pytree flatten: dict keys sorted; tuples in order)
#define P_ATTN_W(l)  P[0 + 2*(l)]
#define P_ATTN_B(l)  P[1 + 2*(l)]
#define P_CCOMP_W(l) P[6 + 2*(l)]
#define P_CCOMP_B(l) P[7 + 2*(l)]
#define P_EMB_W      P[12]
#define P_EMB_B      P[13]
#define P_FIN_W      P[14]
#define P_FIN_B      P[15]
#define P_GAT_W(l)   P[16 + 6*(l)]
#define P_GAT_WE(l)  P[17 + 6*(l)]
#define P_GAT_AD(l)  P[18 + 6*(l)]
#define P_GAT_AE(l)  P[19 + 6*(l)]
#define P_GAT_AS(l)  P[20 + 6*(l)]
#define P_GAT_B(l)   P[21 + 6*(l)]
#define P_SELF_W(l)  P[34 + 2*(l)]
#define P_SELF_B(l)  P[35 + 2*(l)]
#define P_XCOMP_W(l) P[40 + 2*(l)]
#define P_XCOMP_B(l) P[41 + 2*(l)]

struct Scratch {
  float *hB, *hC, *hA, *msg;
  float *a_src, *a_dst, *m, *den, *alpha, *ex, *ws, *wd, *we, *lat, *z;
  unsigned short *hAbf, *hBbf, *hCbf, *commbf, *phbf, *wt;
};

template<int K, int O>
static inline void launch_gemm(const unsigned short* A, const unsigned short* WT,
                               const float* bias, const float* gate, float* C,
                               unsigned short* Cbf, int ldC, int colOff,
                               long aBS, long cBS, int T, int amapDiv, int epi,
                               hipStream_t stream) {
  constexpr int colGroups = O / 64;
  constexpr int nTiles = (NN / 16) * colGroups;
  dim3 grid(nTiles / 8, T);
  gemm_wmma<K, O><<<grid, 256, 0, stream>>>(A, WT, bias, gate, C, Cbf,
                                            ldC, colOff, aBS, cBS, amapDiv, epi);
}

static inline void launch_wconv(const float* W, unsigned short* WT, int K, int O, int T,
                                hipStream_t stream) {
  long n = (long)T * K * O;
  wconv_kernel<<<(int)((n + 255) / 256), 256, 0, stream>>>(W, WT, K, O, n);
}

static void run_encoder(const float* const* P, const float* xin, int inDim,
                        float* finalOut, int outDim, const int* ei, const float* ea,
                        const Scratch& S, hipStream_t stream) {
  const long hbs128 = (long)NN * HH;
  long nEmb = (long)N_TYPES * NN * EMBD;
  emb_kernel<<<(int)((nEmb + 255) / 256), 256, 0, stream>>>(xin, P_EMB_W, P_EMB_B, S.hAbf, inDim);

  for (int l = 0; l < 3; ++l) {
    const int Kcur = (l == 0) ? EMBD : HH;
    const long hbs = (long)NN * Kcur;
    // h = leaky(lin(h, self)) -> hB (f32 for node_att) + hBbf (for ph/xcomp)
    launch_wconv(P_SELF_W(l), S.wt, Kcur, HH, N_TYPES, stream);
    if (l == 0)
      launch_gemm<EMBD, HH>(S.hAbf, S.wt, P_SELF_B(l), nullptr, S.hB, S.hBbf,
                            HH, 0, hbs, hbs128, N_TYPES, 1, 1, stream);
    else
      launch_gemm<HH, HH>(S.hAbf, S.wt, P_SELF_B(l), nullptr, S.hB, S.hBbf,
                          HH, 0, hbs, hbs128, N_TYPES, 1, 1, stream);
    // attention vectors (replaces pd GEMM + ph reductions)
    attvec_kernel<<<N_ETYPES, HH, 0, stream>>>(P_GAT_W(l), P_GAT_AS(l), P_GAT_AD(l),
                                               P_GAT_WE(l), P_GAT_AE(l), S.ws, S.wd, S.we);
    // ph = h[ST] x W (25 batches) -> bf16 only
    launch_wconv(P_GAT_W(l), S.wt, HH, HH, N_ETYPES, stream);
    launch_gemm<HH, HH>(S.hBbf, S.wt, nullptr, nullptr, nullptr, S.phbf,
                        HH, 0, hbs128, hbs128, N_ETYPES, N_TYPES, 0, stream);
    // per-node attention scalars
    long nna = (long)N_ETYPES * NN;
    node_att_kernel<<<(int)((nna + 255) / 256), 256, 0, stream>>>(S.hB, S.ws, S.wd,
                                                                  S.a_src, S.a_dst);
    // edge softmax + scatter aggregate
    fill_kernel<<<(int)((nna + 255) / 256), 256, 0, stream>>>(S.m, -INFINITY, nna);
    fill_kernel<<<(int)((nna + 255) / 256), 256, 0, stream>>>(S.den, 0.f, nna);
    long nmsg = (long)N_ETYPES * NN * HH;
    fill_kernel<<<(int)((nmsg + 255) / 256), 256, 0, stream>>>(S.msg, 0.f, nmsg);
    long nedge = (long)N_ETYPES * EE;
    edge_alpha_kernel<<<(int)((nedge + 255) / 256), 256, 0, stream>>>(ei, ea, S.we, S.a_src,
                                                                      S.a_dst, S.alpha, S.m);
    edge_exp_kernel<<<(int)((nedge + 255) / 256), 256, 0, stream>>>(ei, S.alpha, S.m,
                                                                    S.ex, S.den);
    long nmt = nedge * 32;
    edge_msg_kernel<<<(int)((nmt + 255) / 256), 256, 0, stream>>>(ei, S.ex, S.den,
                                                                  S.phbf, S.msg);
    // cross-source-type max (+bias) + leaky -> commbf
    long ncm = (long)N_TYPES * NN * HH;
    comm_kernel<<<(int)((ncm + 255) / 256), 256, 0, stream>>>(S.msg, P_GAT_B(l), S.commbf);
    // hC = concat(lin(h, xcomp), lin(comm, ccomp))  (f32 gate src + bf16)
    launch_wconv(P_XCOMP_W(l), S.wt, HH, HH / 2, N_TYPES, stream);
    launch_gemm<HH, HH / 2>(S.hBbf, S.wt, P_XCOMP_B(l), nullptr, S.hC, S.hCbf,
                            HH, 0, hbs128, hbs128, N_TYPES, 1, 0, stream);
    launch_wconv(P_CCOMP_W(l), S.wt, HH, HH / 2, N_TYPES, stream);
    launch_gemm<HH, HH / 2>(S.commbf, S.wt, P_CCOMP_B(l), nullptr, S.hC, S.hCbf,
                            HH, HH / 2, hbs128, hbs128, N_TYPES, 1, 0, stream);
    // h = hC + sigmoid(lin(hC, attn))*hC  -> hA (f32) + hAbf (next layer input)
    launch_wconv(P_ATTN_W(l), S.wt, HH, HH, N_TYPES, stream);
    launch_gemm<HH, HH>(S.hCbf, S.wt, P_ATTN_B(l), S.hC, S.hA, S.hAbf,
                        HH, 0, hbs128, hbs128, N_TYPES, 1, 2, stream);
  }
  long nf = (long)N_TYPES * NN * outDim;
  final_kernel<<<(int)((nf + 255) / 256), 256, 0, stream>>>(S.hA, P_FIN_W, P_FIN_B,
                                                            finalOut, outDim);
}

extern "C" void kernel_launch(void* const* d_in, const int* in_sizes, int n_in,
                              void* d_out, int out_size, void* d_ws, size_t ws_size,
                              hipStream_t stream) {
  (void)in_sizes; (void)n_in; (void)out_size; (void)ws_size;
  const float* x   = (const float*)d_in[0];
  const float* eps = (const float*)d_in[1];
  const int*   ei  = (const int*)d_in[2];
  const float* ea  = (const float*)d_in[3];
  const float* encP[46]; const float* decP[46];
  for (int i = 0; i < 46; ++i) {
    encP[i] = (const float*)d_in[4 + i];
    decP[i] = (const float*)d_in[50 + i];
  }

  char* base = (char*)d_ws;
  auto takeB = [&](size_t bytes) {
    char* r = base; base += (bytes + 255) & ~(size_t)255; return r;
  };
  Scratch S;
  S.hA     = (float*)takeB((size_t)N_TYPES * NN * HH * 4);
  S.hB     = (float*)takeB((size_t)N_TYPES * NN * HH * 4);
  S.hC     = (float*)takeB((size_t)N_TYPES * NN * HH * 4);
  S.msg    = (float*)takeB((size_t)N_ETYPES * NN * HH * 4);
  S.a_src  = (float*)takeB((size_t)N_ETYPES * NN * 4);
  S.a_dst  = (float*)takeB((size_t)N_ETYPES * NN * 4);
  S.m      = (float*)takeB((size_t)N_ETYPES * NN * 4);
  S.den    = (float*)takeB((size_t)N_ETYPES * NN * 4);
  S.alpha  = (float*)takeB((size_t)N_ETYPES * EE * 4);
  S.ex     = (float*)takeB((size_t)N_ETYPES * EE * 4);
  S.ws     = (float*)takeB((size_t)N_ETYPES * HH * 4);
  S.wd     = (float*)takeB((size_t)N_ETYPES * HH * 4);
  S.we     = (float*)takeB(512);
  S.lat    = (float*)takeB((size_t)N_TYPES * NN * 2 * ZDIM * 4);
  S.z      = (float*)takeB((size_t)N_TYPES * NN * ZDIM * 4);
  S.hAbf   = (unsigned short*)takeB((size_t)N_TYPES * NN * HH * 2);
  S.hBbf   = (unsigned short*)takeB((size_t)N_TYPES * NN * HH * 2);
  S.hCbf   = (unsigned short*)takeB((size_t)N_TYPES * NN * HH * 2);
  S.commbf = (unsigned short*)takeB((size_t)N_TYPES * NN * HH * 2);
  S.phbf   = (unsigned short*)takeB((size_t)N_ETYPES * NN * HH * 2);
  S.wt     = (unsigned short*)takeB((size_t)N_ETYPES * HH * HH * 2);

  float* out = (float*)d_out;                       // decoder output (5,4096,3)
  float* muO = out + (long)N_TYPES * NN * TDIM;     // mu
  float* lvO = muO + (long)N_TYPES * NN * ZDIM;     // logvar

  // encoder -> lat (mu,logvar interleaved)
  run_encoder(encP, x, TDIM, S.lat, 2 * ZDIM, ei, ea, S, stream);
  // reparameterize
  long nz = (long)N_TYPES * NN * ZDIM;
  reparam_kernel<<<(int)((nz + 255) / 256), 256, 0, stream>>>(S.lat, eps, S.z, muO, lvO);
  // decoder -> out
  run_encoder(decP, S.z, ZDIM, out, TDIM, ei, ea, S, stream);
}
